// SpellerModel_18021682774494
// MI455X (gfx1250) — compile-verified, matrix operand order checked by reading, not run on hardware
//
#include <hip/hip_runtime.h>
#include <hip/hip_bf16.h>
#include <math.h>

// Speller greedy-decode, MI455X (gfx1250, wave32, WMMA bf16 16x16x32, fp32 acc).
// ctx is identically zero in the reference -> CTX columns of Wih0/W1 dropped.
// Per step: 3x fused (gate GEMM + LSTM cell) kernels + 1 fused head kernel.
// Gate kernel: wave owns 16 batch rows x 16 hidden units x all 4 gates
// (acc[t] = gate t at n = t*256 + j), so the cell is a per-lane epilogue.
// W tile staged in LDS (double buffered) via gfx1250 async global->LDS.

#define Hh   256
#define Bb   1024
#define Vv   33
#define Ll   3
#define G4   1024   // 4*H
#define KC   512    // H (input) + H (hidden)
#define CTX  128

typedef __bf16 bf16;
typedef __attribute__((ext_vector_type(16))) __bf16 v16bf;
typedef __attribute__((ext_vector_type(8)))  __bf16 v8bf;
typedef __attribute__((ext_vector_type(8)))  float  v8f;
typedef __attribute__((ext_vector_type(4)))  int    v4i;

#if __has_builtin(__builtin_amdgcn_global_load_async_to_lds_b128) && \
    __has_builtin(__builtin_amdgcn_s_wait_asynccnt)
#define ASYNC_LDS 1
#endif

static __device__ __forceinline__ float sigf(float x) { return 1.0f / (1.0f + __expf(-x)); }

// Stage one 16-byte chunk of W into LDS.
static __device__ __forceinline__ void stage16(const bf16* g, bf16* l) {
#ifdef ASYNC_LDS
    __builtin_amdgcn_global_load_async_to_lds_b128(
        (__attribute__((address_space(1))) v4i*)g,
        (__attribute__((address_space(3))) v4i*)l, 0, 0);
#else
    *(v8bf*)l = *(const v8bf*)g;
#endif
}
static __device__ __forceinline__ void stage_fence() {
#ifdef ASYNC_LDS
    __builtin_amdgcn_s_wait_asynccnt(0);
#endif
    __syncthreads();
}

// ---------------- prep kernels (cheap, deterministic, run every call) ----------------

__global__ void prep_wcat(const float* __restrict__ Wih0, const float* __restrict__ Wih_rest,
                          const float* __restrict__ Whh, bf16* __restrict__ Wcat) {
    int i = blockIdx.x * blockDim.x + threadIdx.x;
    if (i >= Ll * G4 * KC) return;
    int l = i / (G4 * KC);
    int r = i % (G4 * KC);
    int n = r / KC;
    int k = r % KC;
    float v;
    if (k < Hh) {
        v = (l == 0) ? Wih0[n * (Hh + CTX) + k]
                     : Wih_rest[(size_t)(l - 1) * G4 * Hh + (size_t)n * Hh + k];
    } else {
        v = Whh[(size_t)l * G4 * Hh + (size_t)n * Hh + (k - Hh)];
    }
    Wcat[i] = (bf16)v;
}

__global__ void prep_small(const float* __restrict__ W1, const float* __restrict__ embed,
                           const float* __restrict__ bih, const float* __restrict__ bhh,
                           bf16* __restrict__ W1b, bf16* __restrict__ embed_b,
                           float* __restrict__ biascat) {
    int i = blockIdx.x * blockDim.x + threadIdx.x;
    if (i < Hh * Hh) {
        int n = i / Hh, k = i % Hh;
        W1b[i] = (bf16)W1[n * (Hh + CTX) + k];
    }
    if (i < Vv * Hh) embed_b[i] = (bf16)embed[i];
    if (i < Ll * G4) biascat[i] = bih[i] + bhh[i];
}

__global__ void init_state(const float* __restrict__ h0, const float* __restrict__ c0,
                           bf16* __restrict__ hA, float* __restrict__ cbuf,
                           int* __restrict__ charbuf) {
    int i = blockIdx.x * blockDim.x + threadIdx.x;
    if (i < Ll * Bb * Hh) {
        int l = i / (Bb * Hh);
        int j = i % Hh;
        hA[i]   = (bf16)h0[l * Hh + j];
        cbuf[i] = c0[l * Hh + j];
    }
    if (i < Bb) charbuf[i] = 0;
}

// ---------------- fused LSTM layer: gate GEMM + cell epilogue ----------------
// Grid: (256/16 j-tiles, 1024/128 m-blocks); 256 threads = 8 waves, wave = 16 rows.
// acc[t] = 16x16 tile of gate t (W rows t*256 + j0 .. +15), K = 512 = [x | h_prev].
__global__ __launch_bounds__(256)
void lstm_layer_fused(const bf16* __restrict__ A0,      // x rows [B][H], or embed table if gather
                      const int*  __restrict__ gather,  // charbuf for layer 0, else nullptr
                      const bf16* __restrict__ Hprev,   // h_prev[l]  [B][H]
                      const bf16* __restrict__ W,       // Wcat[l]    [1024][512]
                      const float* __restrict__ biasl,  // biascat[l] [1024]
                      float* __restrict__ Cc,           // cbuf[l]    [B][H] in/out
                      bf16* __restrict__ Hout) {        // h_cur[l]   [B][H]
    __shared__ __attribute__((aligned(16))) bf16 Wt[2][4][16][32];   // 8 KB, double buffered

    const int tid  = threadIdx.x;
    const int lane = tid & 31, wave = tid >> 5;
    const int j0   = blockIdx.x * 16;
    const int m0   = blockIdx.y * 128 + wave * 16;
    const int halfsel = lane >> 4;         // ISA 7.12.2: lane<16 -> K{0..7,16..23}, lane>=16 -> K{8..15,24..31}
    const int koffb   = halfsel * 8;
    const int jlane   = lane & 15;
    const int arow    = m0 + jlane;

    const bf16* a0row = gather ? (A0 + (size_t)gather[arow] * Hh)
                               : (A0 + (size_t)arow * Hh);
    const bf16* a1row = Hprev + (size_t)arow * Hh;

    // cooperative W stage: thread -> (gate, row, k-chunk); one b128 each covers 4x16x32 bf16
    const int lg = tid >> 6, lr = (tid >> 2) & 15, lk = tid & 3;
    const bf16* wsrc = W + (size_t)(lg * Hh + j0 + lr) * KC + lk * 8;
    bf16* ldst0 = &Wt[0][lg][lr][lk * 8];
    bf16* ldst1 = &Wt[1][lg][lr][lk * 8];

    stage16(wsrc, ldst0);
    stage_fence();

    v8f acc[4] = {};
    int p = 0;
    for (int kk = 0; kk < KC; kk += 32) {
        if (kk + 32 < KC) stage16(wsrc + kk + 32, p ? ldst0 : ldst1);

        const bf16* ar = (kk < Hh) ? (a0row + kk) : (a1row + (kk - Hh));
        v8bf alo = *(const v8bf*)(ar + koffb);
        v8bf ahi = *(const v8bf*)(ar + 16 + koffb);
        v16bf a;
#pragma unroll
        for (int i = 0; i < 8; i++) { a[i] = alo[i]; a[i + 8] = ahi[i]; }

#pragma unroll
        for (int t = 0; t < 4; t++) {
            v8bf blo = *(const v8bf*)&Wt[p][t][jlane][koffb];
            v8bf bhi = *(const v8bf*)&Wt[p][t][jlane][16 + koffb];
            v16bf bm;
#pragma unroll
            for (int i = 0; i < 8; i++) { bm[i] = blo[i]; bm[i + 8] = bhi[i]; }
            acc[t] = __builtin_amdgcn_wmma_f32_16x16x32_bf16(
                false, a, false, bm, (short)0, acc[t], false, false);
        }
        stage_fence();
        p ^= 1;
    }

    // fused LSTM cell: lane holds i/f/g/o for (rows mbase..mbase+7, unit j)
    const int j = j0 + jlane;
    const int mbase = m0 + halfsel * 8;
    const float bi  = biasl[j];
    const float bf_ = biasl[Hh + j];
    const float bg  = biasl[2 * Hh + j];
    const float bo  = biasl[3 * Hh + j];
#pragma unroll
    for (int r = 0; r < 8; r++) {
        size_t ci = (size_t)(mbase + r) * Hh + j;
        float gi = acc[0][r] + bi;
        float gf = acc[1][r] + bf_;
        float gg = acc[2][r] + bg;
        float go = acc[3][r] + bo;
        float c  = Cc[ci];
        float c2 = sigf(gf) * c + sigf(gi) * tanhf(gg);
        float h2 = sigf(go) * tanhf(c2);
        Cc[ci]   = c2;
        Hout[ci] = (bf16)h2;
    }
}

// ---------------- fused head: FC1 (WMMA + ReLU) -> LDS -> W2 + argmax ----------------
// Grid: 32 WGs x 256 threads; WG covers 32 batch rows x all 256 FC1 outputs.
__global__ __launch_bounds__(256)
void head_fused(const bf16* __restrict__ Hin, const bf16* __restrict__ W1b,
                const float* __restrict__ b1, const float* __restrict__ W2,
                const float* __restrict__ b2, float* __restrict__ out,
                int* __restrict__ charbuf, int t, int Tn) {
    __shared__ float hid[32][Hh];         // 32 KB
    __shared__ float preds[32][Vv + 1];

    const int tid  = threadIdx.x;
    const int lane = tid & 31, wave = tid >> 5;
    const int halfsel = lane >> 4, koffb = halfsel * 8, jlane = lane & 15;
    const int b0    = blockIdx.x * 32;
    const int mloc0 = (wave >> 2) * 16;   // 0 or 16
    const int n0    = (wave & 3) * 64;    // 0,64,128,192
    const bf16* arp = Hin + (size_t)(b0 + mloc0 + jlane) * Hh;

    v8f acc[4] = {};
    for (int kk = 0; kk < Hh; kk += 32) {
        v8bf alo = *(const v8bf*)(arp + kk + koffb);
        v8bf ahi = *(const v8bf*)(arp + kk + 16 + koffb);
        v16bf a;
#pragma unroll
        for (int i = 0; i < 8; i++) { a[i] = alo[i]; a[i + 8] = ahi[i]; }
#pragma unroll
        for (int t4 = 0; t4 < 4; t4++) {
            const bf16* wr = W1b + (size_t)(n0 + t4 * 16 + jlane) * Hh + kk;
            v8bf blo = *(const v8bf*)(wr + koffb);
            v8bf bhi = *(const v8bf*)(wr + 16 + koffb);
            v16bf bm;
#pragma unroll
            for (int i = 0; i < 8; i++) { bm[i] = blo[i]; bm[i + 8] = bhi[i]; }
            acc[t4] = __builtin_amdgcn_wmma_f32_16x16x32_bf16(
                false, a, false, bm, (short)0, acc[t4], false, false);
        }
    }
#pragma unroll
    for (int t4 = 0; t4 < 4; t4++) {
        int n = n0 + t4 * 16 + jlane;
        float bb = b1[n];
#pragma unroll
        for (int r = 0; r < 8; r++) {
            hid[mloc0 + halfsel * 8 + r][n] = fmaxf(acc[t4][r] + bb, 0.0f);
        }
    }
    __syncthreads();

    for (int idx = tid; idx < 32 * Vv; idx += 256) {
        int row = idx / Vv, v = idx % Vv;
        const float* w = W2 + (size_t)v * Hh;
        float s = b2[v];
#pragma unroll 4
        for (int k = 0; k < Hh; k++) s += hid[row][k] * w[k];
        preds[row][v] = s;
        out[(size_t)(b0 + row) * Tn * Vv + (size_t)t * Vv + v] = s;
    }
    __syncthreads();

    if (tid < 32) {
        int best = 0;
        float bv = preds[tid][0];
        for (int k = 1; k < Vv; k++)
            if (preds[tid][k] > bv) { bv = preds[tid][k]; best = k; }
        charbuf[b0 + tid] = best;
    }
}

// ---------------- host orchestration ----------------

extern "C" void kernel_launch(void* const* d_in, const int* in_sizes, int n_in,
                              void* d_out, int out_size, void* d_ws, size_t ws_size,
                              hipStream_t stream) {
    const float* embed    = (const float*)d_in[1];
    const float* Wih0     = (const float*)d_in[2];
    const float* Wih_rest = (const float*)d_in[3];
    const float* Whh      = (const float*)d_in[4];
    const float* bih      = (const float*)d_in[5];
    const float* bhh      = (const float*)d_in[6];
    const float* W1       = (const float*)d_in[7];
    const float* b1       = (const float*)d_in[8];
    const float* W2       = (const float*)d_in[9];
    const float* b2       = (const float*)d_in[10];
    const float* h0       = (const float*)d_in[11];
    const float* c0       = (const float*)d_in[12];
    float* out = (float*)d_out;

    const int Tn = in_sizes[0] / Bb;   // labels is [B, T]

    char*  ws  = (char*)d_ws;
    size_t off = 0;
    auto alloc = [&](size_t bytes) -> char* {
        char* p = ws + off;
        off = (off + bytes + 255) & ~(size_t)255;
        return p;
    };
    bf16*  Wcat    = (bf16*)alloc((size_t)Ll * G4 * KC * sizeof(bf16));
    bf16*  W1b     = (bf16*)alloc((size_t)Hh * Hh * sizeof(bf16));
    bf16*  embed_b = (bf16*)alloc((size_t)Vv * Hh * sizeof(bf16));
    float* biascat = (float*)alloc((size_t)Ll * G4 * sizeof(float));
    bf16*  hA      = (bf16*)alloc((size_t)Ll * Bb * Hh * sizeof(bf16));
    bf16*  hB      = (bf16*)alloc((size_t)Ll * Bb * Hh * sizeof(bf16));
    float* cbuf    = (float*)alloc((size_t)Ll * Bb * Hh * sizeof(float));
    int*   charbuf = (int*)alloc((size_t)Bb * sizeof(int));
    (void)ws_size; (void)n_in; (void)out_size;

    prep_wcat<<<(Ll * G4 * KC + 255) / 256, 256, 0, stream>>>(Wih0, Wih_rest, Whh, Wcat);
    prep_small<<<(Hh * Hh + 255) / 256, 256, 0, stream>>>(W1, embed, bih, bhh, W1b, embed_b, biascat);
    init_state<<<(Ll * Bb * Hh + 255) / 256, 256, 0, stream>>>(h0, c0, hA, cbuf, charbuf);

    const dim3 gridGate(Hh / 16, Bb / 128);   // 16 x 8 = 128 WGs
    for (int t = 0; t < Tn; t++) {
        bf16* hRead  = (t & 1) ? hB : hA;     // h(t-1)
        bf16* hWrite = (t & 1) ? hA : hB;     // h(t)
        for (int l = 0; l < Ll; l++) {
            const bf16* A0  = (l == 0) ? embed_b : (hWrite + (size_t)(l - 1) * Bb * Hh);
            const int*  gth = (l == 0) ? charbuf : nullptr;
            lstm_layer_fused<<<gridGate, 256, 0, stream>>>(
                A0, gth, hRead + (size_t)l * Bb * Hh,
                Wcat + (size_t)l * G4 * KC, biascat + l * G4,
                cbuf + (size_t)l * Bb * Hh, hWrite + (size_t)l * Bb * Hh);
        }
        head_fused<<<Bb / 32, 256, 0, stream>>>(hWrite + (size_t)2 * Bb * Hh,
                                                W1b, b1, W2, b2, out, charbuf, t, Tn);
    }
}
